// SpatialBayesianLayer_70987219468594
// MI455X (gfx1250) — 8-bit matrix operand order repaired and checked by reading, compiled, not run on hardware
//
#include <hip/hip_runtime.h>
#include <hip/hip_bf16.h>
#include <math.h>

typedef __attribute__((ext_vector_type(8))) int v8i;

#define EPS_RMS  1e-6f
#define EPS_Q    1e-5f
#define EPS_NORM 1e-10f

#define NB   32
#define NS   1024
#define NF   512
#define M_TOK (NB * NS)
#define K1   (2 * NF)
#define K2   NF

__global__ __launch_bounds__(256) void wmean_kernel(const float* __restrict__ w,
                                                    int n, float* __restrict__ wdeq) {
    __shared__ float sbuf[256];
    float s = 0.0f;
    for (int i = threadIdx.x; i < n; i += 256) s += fabsf(w[i]);
    sbuf[threadIdx.x] = s;
    __syncthreads();
    for (int stride = 128; stride > 0; stride >>= 1) {
        if (threadIdx.x < stride) sbuf[threadIdx.x] += sbuf[threadIdx.x + stride];
        __syncthreads();
    }
    if (threadIdx.x == 0) *wdeq = fmaxf(sbuf[0] / (float)n, EPS_Q);
}

__global__ __launch_bounds__(256) void wpack_kernel(const float* __restrict__ w,
                                                    const float* __restrict__ wdeqp,
                                                    signed char* __restrict__ q, int n) {
    int i = blockIdx.x * 256 + threadIdx.x;
    if (i >= n) return;
    float scale = 1.0f / *wdeqp;
    float t = fminf(fmaxf(rintf(w[i] * scale), -1.0f), 1.0f);
    q[i] = (signed char)t;
}

__global__ __launch_bounds__(256) void act1_quant_kernel(const float* __restrict__ prior,
                                                         const float* __restrict__ evid,
                                                         signed char* __restrict__ xq1,
                                                         float* __restrict__ inv_a1) {
    int tok  = blockIdx.x * 8 + (threadIdx.x >> 5);
    int lane = threadIdx.x & 31;
    if (tok >= M_TOK) return;
    const float* p = prior + (size_t)tok * NF;
    const float* e = evid  + (size_t)tok * NF;

    float v[32];
    float ss = 0.0f;
    #pragma unroll
    for (int i = 0; i < 32; ++i) {
        int j = lane + 32 * i;
        float x = (i < 16) ? p[j] : e[j - NF];
        v[i] = x;
        ss += x * x;
    }
    #pragma unroll
    for (int m = 16; m >= 1; m >>= 1) ss += __shfl_xor(ss, m, 32);
    float rms = rsqrtf(ss * (1.0f / (float)K1) + EPS_RMS);

    float amax = 0.0f;
    #pragma unroll
    for (int i = 0; i < 32; ++i) { v[i] *= rms; amax = fmaxf(amax, fabsf(v[i])); }
    #pragma unroll
    for (int m = 16; m >= 1; m >>= 1) amax = fmaxf(amax, __shfl_xor(amax, m, 32));
    float scale = 127.0f / fmaxf(amax, EPS_Q);

    signed char* orow = xq1 + (size_t)tok * K1;
    #pragma unroll
    for (int i = 0; i < 32; ++i) {
        float q = fminf(fmaxf(rintf(v[i] * scale), -128.0f), 127.0f);
        orow[lane + 32 * i] = (signed char)q;
    }
    if (lane == 0) inv_a1[tok] = 1.0f / scale;
}

__device__ __forceinline__ void async_stage_to_lds(const signed char* gbase,
                                                   signed char* lds_ptr, int nbytes) {
    unsigned lds_base = (unsigned)(uintptr_t)lds_ptr;
    for (int t = threadIdx.x; t < (nbytes >> 4); t += 256) {
        unsigned lds_addr = lds_base + (unsigned)(t << 4);
        const signed char* g = gbase + ((size_t)t << 4);
        asm volatile("global_load_async_to_lds_b128 %0, %1, off"
                     :: "v"(lds_addr), "v"(g) : "memory");
    }
    asm volatile("s_wait_asynccnt 0x0" ::: "memory");
}

__device__ __forceinline__ v8i load_fragA(const signed char* arow, int kb) {
    v8i a;
    #pragma unroll
    for (int j = 0; j < 8; ++j) {
        int koff = kb + (j & 3) * 4 + (j >> 2) * 32;
        a[j] = *(const int*)(arow + koff);
    }
    return a;
}
__device__ __forceinline__ v8i load_fragB(const signed char* brow, int kb) {
    v8i b;
    #pragma unroll
    for (int j = 0; j < 8; ++j) {
        int koff = kb + (j & 3) * 4 + (j >> 2) * 32;
        b[j] = *(const int*)(brow + koff);
    }
    return b;
}

__global__ __launch_bounds__(256) void gemm1_fused_kernel(const signed char* __restrict__ xq,
                                                          const signed char* __restrict__ wq,
                                                          const float* __restrict__ inv_a1,
                                                          const float* __restrict__ wdeqp,
                                                          const float* __restrict__ bias,
                                                          signed char* __restrict__ xq2,
                                                          float* __restrict__ inv_a2) {
    __shared__ signed char a_sh[16 * K1];
    __shared__ float hrow[16][NF];
    const int mbase = blockIdx.x * 16;
    const int lane  = threadIdx.x & 31;
    const int wave  = threadIdx.x >> 5;
    const int ln    = lane & 15;
    const bool hi   = lane >= 16;
    const float wdeq = *wdeqp;

    async_stage_to_lds(xq + (size_t)mbase * K1, a_sh, 16 * K1);
    __syncthreads();

    const signed char* arow = a_sh + ln * K1 + (hi ? 16 : 0);

    v8i acc[4] = {};
    for (int kb = 0; kb < K1; kb += 64) {
        v8i a = load_fragA(arow, kb);
        #pragma unroll
        for (int nt = 0; nt < 4; ++nt) {
            int n = (wave * 4 + nt) * 16 + ln;
            const signed char* brow = wq + (size_t)n * K1 + (hi ? 16 : 0);
            if (kb + 64 < K1) __builtin_prefetch(brow + kb + 64, 0, 3);
            v8i b = load_fragB(brow, kb);
            acc[nt] = __builtin_amdgcn_wmma_i32_16x16x64_iu8(
                true, a, true, b, acc[nt], false, false);
        }
    }

    #pragma unroll
    for (int nt = 0; nt < 4; ++nt) {
        int n = (wave * 4 + nt) * 16 + ln;
        float bn = bias[n];
        #pragma unroll
        for (int r = 0; r < 8; ++r) {
            int ml = r + (hi ? 8 : 0);
            float y = (float)acc[nt][r] * inv_a1[mbase + ml] * wdeq + bn;
            hrow[ml][n] = fmaxf(y, 0.0f);
        }
    }
    __syncthreads();

    #pragma unroll
    for (int rr = 0; rr < 2; ++rr) {
        int ml = wave * 2 + rr;
        float v[16];
        float ss = 0.0f;
        #pragma unroll
        for (int i = 0; i < 16; ++i) { v[i] = hrow[ml][lane + 32 * i]; ss += v[i] * v[i]; }
        #pragma unroll
        for (int m = 16; m >= 1; m >>= 1) ss += __shfl_xor(ss, m, 32);
        float rms = rsqrtf(ss * (1.0f / (float)K2) + EPS_RMS);
        float amax = 0.0f;
        #pragma unroll
        for (int i = 0; i < 16; ++i) { v[i] *= rms; amax = fmaxf(amax, fabsf(v[i])); }
        #pragma unroll
        for (int m = 16; m >= 1; m >>= 1) amax = fmaxf(amax, __shfl_xor(amax, m, 32));
        float scale = 127.0f / fmaxf(amax, EPS_Q);
        signed char* orow = xq2 + (size_t)(mbase + ml) * K2;
        #pragma unroll
        for (int i = 0; i < 16; ++i) {
            float q = fminf(fmaxf(rintf(v[i] * scale), -128.0f), 127.0f);
            orow[lane + 32 * i] = (signed char)q;
        }
        if (lane == 0) inv_a2[mbase + ml] = 1.0f / scale;
    }
}

__global__ __launch_bounds__(256) void gemm2_kernel(const signed char* __restrict__ xq,
                                                    const signed char* __restrict__ wq,
                                                    const float* __restrict__ inv_a2,
                                                    const float* __restrict__ wdeqp,
                                                    const float* __restrict__ bias,
                                                    const float* __restrict__ prior,
                                                    float* __restrict__ out) {
    __shared__ signed char a_sh[16 * K2];
    int mt = blockIdx.x >> 2;
    int nt = ((blockIdx.x & 3) << 3) + (threadIdx.x >> 5);
    int mbase = mt * 16, nbase = nt * 16;
    int lane = threadIdx.x & 31;
    int ln = lane & 15;
    bool hi = lane >= 16;
    float wdeq = *wdeqp;

    async_stage_to_lds(xq + (size_t)mbase * K2, a_sh, 16 * K2);
    __syncthreads();

    const signed char* arow = a_sh + ln * K2 + (hi ? 16 : 0);
    const signed char* brow = wq + (size_t)(nbase + ln) * K2 + (hi ? 16 : 0);

    v8i acc = {};
    for (int kb = 0; kb < K2; kb += 64) {
        if (kb + 64 < K2) __builtin_prefetch(brow + kb + 64, 0, 3);
        v8i a = load_fragA(arow, kb);
        v8i b = load_fragB(brow, kb);
        acc = __builtin_amdgcn_wmma_i32_16x16x64_iu8(true, a, true, b, acc, false, false);
    }

    int n = nbase + ln;
    float bn = bias[n];
    #pragma unroll
    for (int r = 0; r < 8; ++r) {
        int m = mbase + r + (hi ? 8 : 0);
        float y = (float)acc[r] * inv_a2[m] * wdeq + bn;
        float sig = 1.0f / (1.0f + expf(-y));
        size_t idx = (size_t)m * NF + n;
        out[idx] = prior[idx] * sig;
    }
}

__global__ __launch_bounds__(256) void colsum_kernel(const float* __restrict__ pu,
                                                     float* __restrict__ sums) {
    int idx = blockIdx.x * 256 + threadIdx.x;
    int b = idx >> 9;
    int f = idx & (NF - 1);
    const float* base = pu + (size_t)b * NS * NF + f;
    float s = 0.0f;
    for (int sI = 0; sI < NS; ++sI) s += base[(size_t)sI * NF];
    sums[idx] = fmaxf(s, EPS_NORM);
}

__global__ __launch_bounds__(256) void normalize_kernel(float* __restrict__ out,
                                                        const float* __restrict__ sums) {
    size_t i = (size_t)blockIdx.x * 256 + threadIdx.x;
    int f = (int)(i & (NF - 1));
    size_t tok = i >> 9;
    int b = (int)(tok >> 10);
    out[i] = out[i] / sums[(b << 9) | f];
}

static inline size_t align256(size_t x) { return (x + 255) & ~(size_t)255; }

extern "C" void kernel_launch(void* const* d_in, const int* in_sizes, int n_in,
                              void* d_out, int out_size, void* d_ws, size_t ws_size,
                              hipStream_t stream) {
    const float* evid  = (const float*)d_in[0];
    const float* prior = (const float*)d_in[1];
    const float* W1    = (const float*)d_in[2];
    const float* b1    = (const float*)d_in[3];
    const float* W2    = (const float*)d_in[4];
    const float* b2    = (const float*)d_in[5];
    float* out = (float*)d_out;

    char* ws = (char*)d_ws;
    size_t off = 0;
    signed char* xq1   = (signed char*)(ws + off); off = align256(off + (size_t)M_TOK * K1);
    signed char* xq2   = (signed char*)(ws + off); off = align256(off + (size_t)M_TOK * K2);
    signed char* w1q   = (signed char*)(ws + off); off = align256(off + (size_t)NF * K1);
    signed char* w2q   = (signed char*)(ws + off); off = align256(off + (size_t)NF * K2);
    float* inv_a1      = (float*)(ws + off);       off = align256(off + (size_t)M_TOK * 4);
    float* inv_a2      = (float*)(ws + off);       off = align256(off + (size_t)M_TOK * 4);
    float* wdeq1       = (float*)(ws + off);       off = align256(off + 4);
    float* wdeq2       = (float*)(ws + off);       off = align256(off + 4);
    float* sums        = (float*)(ws + off);       off = align256(off + (size_t)NB * NF * 4);
    (void)ws_size; (void)in_sizes; (void)n_in; (void)out_size;

    wmean_kernel<<<1, 256, 0, stream>>>(W1, NF * K1, wdeq1);
    wmean_kernel<<<1, 256, 0, stream>>>(W2, NF * K2, wdeq2);
    wpack_kernel<<<(NF * K1 + 255) / 256, 256, 0, stream>>>(W1, wdeq1, w1q, NF * K1);
    wpack_kernel<<<(NF * K2 + 255) / 256, 256, 0, stream>>>(W2, wdeq2, w2q, NF * K2);

    act1_quant_kernel<<<M_TOK / 8, 256, 0, stream>>>(prior, evid, xq1, inv_a1);

    gemm1_fused_kernel<<<M_TOK / 16, 256, 0, stream>>>(xq1, w1q, inv_a1, wdeq1, b1,
                                                       xq2, inv_a2);

    gemm2_kernel<<<(M_TOK / 16) * (NF / 16) / 8, 256, 0, stream>>>(
        xq2, w2q, inv_a2, wdeq2, b2, prior, out);

    colsum_kernel<<<(NB * NF) / 256, 256, 0, stream>>>(out, sums);
    normalize_kernel<<<(int)(((size_t)M_TOK * NF) / 256), 256, 0, stream>>>(out, sums);
}
